// ResidualBlock_52183852646680
// MI455X (gfx1250) — matrix operand order repaired and checked by reading, not compiled, not run on hardware
//
#include <hip/hip_runtime.h>

// ---------------------------------------------------------------------------
// ResidualBlock (BN -> ReLU -> Kron(64x64) x2 -> BN -> ReLU -> Kron x2 -> +x)
// B=16384, FEAT=4096, D=64.  Memory-bound: ~1.41GB HBM traffic (y1 kept f16)
// => ~60us @ 23.3TB/s. Per-sample 64x64 matmuls on v_wmma_f32_16x16x32_f16;
// bias folded into accumulator init; 2 tiles/wave sharing A fragments.
// BN stats are 2-level (8 row-chunk partials -> finalize) for occupancy.
// ---------------------------------------------------------------------------

typedef __attribute__((ext_vector_type(16))) _Float16 v16h;
typedef __attribute__((ext_vector_type(8)))  _Float16 v8h;
typedef __attribute__((ext_vector_type(8)))  float    v8f;

#define FEAT 4096
#define DD   64
#define LS   72          // LDS row stride in f16 elements (144B, 16B-aligned)
#define BN_EPS 1e-3f
#define NCHUNK 8         // row-chunks for 2-level BN statistics

// ---- WMMA fragment loads from LDS (row-major [row][K], stride LS) ----------
// A 16x32 f16 (ISA 7.12.2): lane L -> M = L&15, half = L>>4,
// VGPR0..3: K = half*8 + {0..7}; VGPR4..7: K = 16 + half*8 + {0..7}.
__device__ __forceinline__ v16h frag_a(const _Float16* base, int row0, int kc, int lane) {
  int r    = row0 + (lane & 15);
  int half = lane >> 4;
  const _Float16* p = base + r * LS + kc + half * 8;
  v8h lo = *(const v8h*)p;
  v8h hi = *(const v8h*)(p + 16);
  return __builtin_shufflevector(lo, hi, 0,1,2,3,4,5,6,7,8,9,10,11,12,13,14,15);
}

// B 32x16 f16 stored N-major (base[N][K]): lane L -> N = L&15;
// half = L>>4: K = half*8 + {0..7}, then K = 16 + half*8 + {0..7} (matches A).
__device__ __forceinline__ v16h frag_b(const _Float16* base, int row0, int kc, int lane) {
  int n    = row0 + (lane & 15);
  int half = lane >> 4;
  const _Float16* p = base + n * LS + kc + half * 8;
  v8h lo = *(const v8h*)p;
  v8h hi = *(const v8h*)(p + 16);
  return __builtin_shufflevector(lo, hi, 0,1,2,3,4,5,6,7,8,9,10,11,12,13,14,15);
}

// Bias -> accumulator init. D layout: VGPR v, lane L -> M = v + 8*(L>>4), N = L&15.
__device__ __forceinline__ v8f bias_init(const _Float16* bias, int tm, int tn, int lane) {
  int half = lane >> 4, lm = lane & 15;
  v8f c;
#pragma unroll
  for (int v = 0; v < 8; ++v)
    c[v] = (float)bias[(tm + v + half * 8) * LS + tn + lm];
  return c;
}

// ---- BN statistics, level 1: per-(feature, row-chunk) partial sums ---------
// grid = (FEAT/64, NCHUNK), 256 threads. Thread t: feature blk.x*64 + (t&63),
// rows r = chunk_base + (t>>6) + 4k. Coalesced across the block.
template <typename T>
__global__ void bn_partial_kernel(const T* __restrict__ x,
                                  float* __restrict__ psum,
                                  float* __restrict__ psq,
                                  int rows_per_chunk) {
  __shared__ float ss[256];
  __shared__ float qq[256];
  const int f  = blockIdx.x * 64 + (threadIdx.x & 63);
  const int rb = blockIdx.y * rows_per_chunk;
  const int re = rb + rows_per_chunk;
  float s = 0.f, q = 0.f;
  for (int r = rb + (threadIdx.x >> 6); r < re; r += 4) {
    const T* p = x + (size_t)r * FEAT + f;
    __builtin_prefetch(p + 16 * FEAT, 0, 1);       // global_prefetch_b8
    float v = (float)*p;
    s += v;
    q += v * v;
  }
  ss[threadIdx.x] = s;
  qq[threadIdx.x] = q;
  __syncthreads();
  if (threadIdx.x < 64) {
    float sum = ss[threadIdx.x] + ss[threadIdx.x + 64] +
                ss[threadIdx.x + 128] + ss[threadIdx.x + 192];
    float sq  = qq[threadIdx.x] + qq[threadIdx.x + 64] +
                qq[threadIdx.x + 128] + qq[threadIdx.x + 192];
    int fo = blockIdx.x * 64 + threadIdx.x;
    psum[blockIdx.y * FEAT + fo] = sum;
    psq [blockIdx.y * FEAT + fo] = sq;
  }
}

// ---- BN statistics, level 2: fold chunk partials into mean / rstd ----------
__global__ void bn_finalize_kernel(const float* __restrict__ psum,
                                   const float* __restrict__ psq,
                                   float* __restrict__ mean,
                                   float* __restrict__ rstd,
                                   int rows) {
  const int f = blockIdx.x * 256 + threadIdx.x;
  float s = 0.f, q = 0.f;
#pragma unroll
  for (int c = 0; c < NCHUNK; ++c) {
    s += psum[c * FEAT + f];
    q += psq [c * FEAT + f];
  }
  float invB = 1.0f / (float)rows;
  float m    = s * invB;
  float var  = q * invB - m * m;
  mean[f] = m;
  rstd[f] = rsqrtf(var + BN_EPS);
}

// ---- Fused BN+ReLU -> (lw^T @ t + lb) -> (t' @ rw + rb) [+ residual] -------
// One workgroup (8 waves) per sample; wave w owns tile row tm=w/2 and tile
// cols {2*(w&1), 2*(w&1)+1}: A fragments shared across both output tiles.
template <typename SrcT, typename DstT, bool RES>
__global__ void kron_block_kernel(const SrcT* __restrict__ src,       // [rows, FEAT]
                                  const float* __restrict__ mean,
                                  const float* __restrict__ rstd,
                                  const float* __restrict__ gamma,
                                  const float* __restrict__ beta,
                                  const float* __restrict__ lw,       // (k,o)
                                  const float* __restrict__ lb,       // (o,j)
                                  const float* __restrict__ rw,       // (j,n)
                                  const float* __restrict__ rb,       // (o,n)
                                  const float* __restrict__ residual, // [rows,FEAT]
                                  DstT* __restrict__ dst) {
  __shared__ __align__(16) _Float16 wA1[DD * LS];  // lw^T : A1[o][k] = lw[k][o]
  __shared__ __align__(16) _Float16 wB1[DD * LS];  // t^T  : B1[j][k] = t[k][j]
  __shared__ __align__(16) _Float16 wA2[DD * LS];  // t'   : A2[o][j]
  __shared__ __align__(16) _Float16 wB2[DD * LS];  // rw^T : B2[n][j] = rw[j][n]
  __shared__ __align__(16) _Float16 wLb[DD * LS];  // lb[o][j]
  __shared__ __align__(16) _Float16 wRb[DD * LS];  // rb[o][n]

  const int    sample = blockIdx.x;
  const size_t sbase  = (size_t)sample * FEAT;

  // Stage weights + biases (f32 -> f16; lw/rw transposed for fragment loads).
  for (int i = threadIdx.x; i < FEAT; i += 256) {
    int k = i >> 6, o = i & 63;                    // i = k*64 + o
    wA1[o * LS + k] = (_Float16)lw[i];             // A1[o][k] = lw[k][o]
    wB2[o * LS + k] = (_Float16)rw[i];             // B2[n=o][j=k] = rw[j][n]
    wLb[k * LS + o] = (_Float16)lb[i];             // lb row-major (o,j)
    wRb[k * LS + o] = (_Float16)rb[i];
  }
  // Stage BN'd + ReLU'd sample, transposed (B operand of matmul 1).
  const SrcT* xs = src + sbase;
  for (int i = threadIdx.x; i < FEAT; i += 256) {
    int k = i >> 6, j = i & 63;                    // t[k][j] = xs[i], feature = i
    float h = ((float)xs[i] - mean[i]) * rstd[i] * gamma[i] + beta[i];
    h = fmaxf(h, 0.0f);
    wB1[j * LS + k] = (_Float16)h;                 // B1[j][k]
  }
  __syncthreads();

  const int lane = threadIdx.x & 31;
  const int wave = threadIdx.x >> 5;
  const int half = lane >> 4;
  const int lm   = lane & 15;
  const int tm   = (wave >> 1) * 16;
  const int tn0  = (wave & 1) * 32;
  const int tn1  = tn0 + 16;

  // ---- matmul 1: t'[o][j] = lb[o][j] + sum_k A1[o][k] * B1[j][k] ----
  {
    v8f c0 = bias_init(wLb, tm, tn0, lane);
    v8f c1 = bias_init(wLb, tm, tn1, lane);
    v16h a0  = frag_a(wA1, tm, 0,  lane);
    v16h a1  = frag_a(wA1, tm, 32, lane);
    v16h b00 = frag_b(wB1, tn0, 0,  lane);
    v16h b10 = frag_b(wB1, tn1, 0,  lane);
    v16h b01 = frag_b(wB1, tn0, 32, lane);
    v16h b11 = frag_b(wB1, tn1, 32, lane);
    c0 = __builtin_amdgcn_wmma_f32_16x16x32_f16(false, a0, false, b00, (short)0, c0, false, false);
    c1 = __builtin_amdgcn_wmma_f32_16x16x32_f16(false, a0, false, b10, (short)0, c1, false, false);
    c0 = __builtin_amdgcn_wmma_f32_16x16x32_f16(false, a1, false, b01, (short)0, c0, false, false);
    c1 = __builtin_amdgcn_wmma_f32_16x16x32_f16(false, a1, false, b11, (short)0, c1, false, false);
#pragma unroll
    for (int v = 0; v < 8; ++v) {
      int r = tm + v + half * 8;
      wA2[r * LS + tn0 + lm] = (_Float16)c0[v];    // A operand of matmul 2
      wA2[r * LS + tn1 + lm] = (_Float16)c1[v];
    }
  }
  __syncthreads();

  // ---- matmul 2: t''[o][n] = rb[o][n] + sum_j A2[o][j] * B2[n][j] ----
  {
    v8f c0 = bias_init(wRb, tm, tn0, lane);
    v8f c1 = bias_init(wRb, tm, tn1, lane);
    v16h a0  = frag_a(wA2, tm, 0,  lane);
    v16h a1  = frag_a(wA2, tm, 32, lane);
    v16h b00 = frag_b(wB2, tn0, 0,  lane);
    v16h b10 = frag_b(wB2, tn1, 0,  lane);
    v16h b01 = frag_b(wB2, tn0, 32, lane);
    v16h b11 = frag_b(wB2, tn1, 32, lane);
    c0 = __builtin_amdgcn_wmma_f32_16x16x32_f16(false, a0, false, b00, (short)0, c0, false, false);
    c1 = __builtin_amdgcn_wmma_f32_16x16x32_f16(false, a0, false, b10, (short)0, c1, false, false);
    c0 = __builtin_amdgcn_wmma_f32_16x16x32_f16(false, a1, false, b01, (short)0, c0, false, false);
    c1 = __builtin_amdgcn_wmma_f32_16x16x32_f16(false, a1, false, b11, (short)0, c1, false, false);

    DstT* outp = dst + sbase;
#pragma unroll
    for (int v = 0; v < 8; ++v) {
      int r = tm + v + half * 8;
      float v0 = c0[v], v1 = c1[v];
      if constexpr (RES) {
        v0 += residual[sbase + r * DD + tn0 + lm];
        v1 += residual[sbase + r * DD + tn1 + lm];
      }
      outp[r * DD + tn0 + lm] = (DstT)v0;
      outp[r * DD + tn1 + lm] = (DstT)v1;
    }
  }
}

// ---------------------------------------------------------------------------
extern "C" void kernel_launch(void* const* d_in, const int* in_sizes, int n_in,
                              void* d_out, int out_size, void* d_ws, size_t ws_size,
                              hipStream_t stream) {
  const float* inputs    = (const float*)d_in[0];
  const float* bn0_gamma = (const float*)d_in[1];
  const float* bn0_beta  = (const float*)d_in[2];
  const float* bn1_gamma = (const float*)d_in[3];
  const float* bn1_beta  = (const float*)d_in[4];
  const float* k1_lw = (const float*)d_in[5];
  const float* k1_lb = (const float*)d_in[6];
  const float* k1_rw = (const float*)d_in[7];
  const float* k1_rb = (const float*)d_in[8];
  const float* k2_lw = (const float*)d_in[9];
  const float* k2_lb = (const float*)d_in[10];
  const float* k2_rw = (const float*)d_in[11];
  const float* k2_rb = (const float*)d_in[12];
  float* out = (float*)d_out;

  const int rows = in_sizes[0] / FEAT;             // 16384

  // Workspace: mean0|rstd0|mean1|rstd1 (4xFEAT f32), psum|psq (2*NCHUNK*FEAT
  // f32, reused by both stats passes), then y1 [rows,FEAT] f16.
  float*    mean0 = (float*)d_ws;
  float*    rstd0 = mean0 + FEAT;
  float*    mean1 = rstd0 + FEAT;
  float*    rstd1 = mean1 + FEAT;
  float*    psum  = rstd1 + FEAT;
  float*    psq   = psum + NCHUNK * FEAT;
  _Float16* y1    = (_Float16*)(psq + NCHUNK * FEAT);

  const dim3 sgrid(FEAT / 64, NCHUNK);
  const int  rpc = rows / NCHUNK;

  // 1) BN0 batch statistics over inputs (f32), 2-level.
  bn_partial_kernel<float><<<sgrid, 256, 0, stream>>>(inputs, psum, psq, rpc);
  bn_finalize_kernel<<<FEAT / 256, 256, 0, stream>>>(psum, psq, mean0, rstd0, rows);
  // 2) BN0+ReLU -> Kron block 1 -> y1 (f16).
  kron_block_kernel<float, _Float16, false><<<rows, 256, 0, stream>>>(
      inputs, mean0, rstd0, bn0_gamma, bn0_beta,
      k1_lw, k1_lb, k1_rw, k1_rb, nullptr, y1);
  // 3) BN1 batch statistics over y1 (f16), 2-level.
  bn_partial_kernel<_Float16><<<sgrid, 256, 0, stream>>>(y1, psum, psq, rpc);
  bn_finalize_kernel<<<FEAT / 256, 256, 0, stream>>>(psum, psq, mean1, rstd1, rows);
  // 4) BN1+ReLU -> Kron block 2 -> + inputs -> out (f32).
  kron_block_kernel<_Float16, float, true><<<rows, 256, 0, stream>>>(
      y1, mean1, rstd1, bn1_gamma, bn1_beta,
      k2_lw, k2_lb, k2_rw, k2_rb, inputs, out);
}
